// SelfAttention_39324720562850
// MI455X (gfx1250) — matrix operand order repaired and checked by reading, not compiled, not run on hardware
//
#include <hip/hip_runtime.h>

typedef __attribute__((ext_vector_type(16))) __bf16 v16bf;
typedef __attribute__((ext_vector_type(8)))  __bf16 v8bf;
typedef __attribute__((ext_vector_type(8)))  float  v8f;

#define DIMD 1024
#define SEQN 2048
#define NB   4
#define NROWS (NB * SEQN)   // 8192 total rows

// ---- WMMA fragment loaders (bf16, 16x16x32) ----------------------------
// A (16x32, MxK): lane l: row = l&15, h = l>>4; elements 0..7 = K h*8+0..7,
// elements 8..15 = K 16+h*8+0..7  -> two 16B loads, 16 halfs apart.
__device__ __forceinline__ v16bf load_fragA(const __bf16* p) {
  v8bf lo = *(const v8bf*)(p);
  v8bf hi = *(const v8bf*)(p + 16);
  v16bf r;
#pragma unroll
  for (int i = 0; i < 8; ++i) { r[i] = lo[i]; r[i + 8] = hi[i]; }
  return r;
}
// B (32x16, KxN) stored transposed [N][K]: lane l: col = l&15, h = l>>4;
// elements 0..7 = K h*8+0..7, elements 8..15 = K 16+h*8+0..7 -> two 16B loads.
__device__ __forceinline__ v16bf load_fragB(const __bf16* p) {
  v8bf lo = *(const v8bf*)(p);
  v8bf hi = *(const v8bf*)(p + 16);
  v16bf r;
#pragma unroll
  for (int i = 0; i < 8; ++i) { r[i] = lo[i]; r[i + 8] = hi[i]; }
  return r;
}

// ---- elementwise converts ----------------------------------------------
__global__ void k_cvt_bf16(const float* __restrict__ in, __bf16* __restrict__ out, int n) {
  int i = blockIdx.x * blockDim.x + threadIdx.x;
  if (i < n) out[i] = (__bf16)in[i];
}

// in: [1024][1024] f32 row-major; out[j][d] = in[d][j] (bf16, transposed)
__global__ void k_transpose_bf16(const float* __restrict__ in, __bf16* __restrict__ out) {
  int i = blockIdx.x * blockDim.x + threadIdx.x;
  int d = i >> 10, j = i & 1023;
  out[(size_t)j * DIMD + d] = (__bf16)in[i];
}

// ---- fused Q/K/V projection GEMM, 32x16 tile per wave ------------------
// C = xb[8192x1024] * W{q,k,v}T ; 2 A frags reused across 3 weights: 6 wmma/kstep.
__global__ void k_qkv(const __bf16* __restrict__ xb,
                      const __bf16* __restrict__ WqT,
                      const __bf16* __restrict__ WkT,
                      const __bf16* __restrict__ WvT,
                      __bf16* __restrict__ qb, __bf16* __restrict__ kb,
                      float* __restrict__ vf, __bf16* __restrict__ vt) {
  const int lane = threadIdx.x & 31;
  const int wid  = (blockIdx.x * blockDim.x + threadIdx.x) >> 5;
  const int tn  = wid & 63;   // 64 col tiles (16 wide)
  const int tm2 = wid >> 6;   // 256 row super-tiles (32 rows)
  const int idx = lane & 15;
  const int h   = lane >> 4;
  const __bf16* arow0 = xb + (size_t)(tm2 * 32 + idx)      * DIMD + h * 8;
  const __bf16* arow1 = xb + (size_t)(tm2 * 32 + 16 + idx) * DIMD + h * 8;
  const size_t  boff = (size_t)(tn * 16 + idx) * DIMD + h * 8;
  const __bf16* bq = WqT + boff;
  const __bf16* bk = WkT + boff;
  const __bf16* bv = WvT + boff;
  v8f cq0 = {}, ck0 = {}, cv0 = {};
  v8f cq1 = {}, ck1 = {}, cv1 = {};
  for (int k0 = 0; k0 < DIMD; k0 += 32) {
    v16bf a0 = load_fragA(arow0 + k0);
    v16bf a1 = load_fragA(arow1 + k0);
    v16bf wq = load_fragB(bq + k0);
    v16bf wk = load_fragB(bk + k0);
    v16bf wv = load_fragB(bv + k0);
    cq0 = __builtin_amdgcn_wmma_f32_16x16x32_bf16(false, a0, false, wq, (short)0, cq0, false, false);
    ck0 = __builtin_amdgcn_wmma_f32_16x16x32_bf16(false, a0, false, wk, (short)0, ck0, false, false);
    cv0 = __builtin_amdgcn_wmma_f32_16x16x32_bf16(false, a0, false, wv, (short)0, cv0, false, false);
    cq1 = __builtin_amdgcn_wmma_f32_16x16x32_bf16(false, a1, false, wq, (short)0, cq1, false, false);
    ck1 = __builtin_amdgcn_wmma_f32_16x16x32_bf16(false, a1, false, wk, (short)0, ck1, false, false);
    cv1 = __builtin_amdgcn_wmma_f32_16x16x32_bf16(false, a1, false, wv, (short)0, cv1, false, false);
  }
  const int col = tn * 16 + idx;
#pragma unroll
  for (int sm = 0; sm < 2; ++sm) {
    const v8f& cq = sm ? cq1 : cq0;
    const v8f& ck = sm ? ck1 : ck0;
    const v8f& cv = sm ? cv1 : cv0;
#pragma unroll
    for (int r = 0; r < 8; ++r) {
      int row = tm2 * 32 + sm * 16 + h * 8 + r;
      size_t o = (size_t)row * DIMD + col;
      qb[o] = (__bf16)cq[r];
      kb[o] = (__bf16)ck[r];
      vf[o] = cv[r];
      // V transposed per batch: vt[b][col][row_in_batch]
      vt[((size_t)(row >> 11) * DIMD + col) * SEQN + (row & 2047)] = (__bf16)cv[r];
    }
  }
}

// ---- attention scores (one batch): S = scale * Q K^T, causal masked ----
// 32x32 super-tile per wave: 2 A + 2 B frags -> 4 wmma/kstep.
__global__ void k_scores(const __bf16* __restrict__ qb, const __bf16* __restrict__ kb,
                         float* __restrict__ S) {
  const int lane = threadIdx.x & 31;
  const int wid  = (blockIdx.x * blockDim.x + threadIdx.x) >> 5;
  const int tj2 = wid & 63, ti2 = wid >> 6;     // 64x64 super-tiles of 32x32
  const int idx = lane & 15, h = lane >> 4;
  if (tj2 > ti2) {                               // fully above diagonal: no GEMM
#pragma unroll
    for (int sm = 0; sm < 2; ++sm)
#pragma unroll
      for (int sn = 0; sn < 2; ++sn)
#pragma unroll
        for (int r = 0; r < 8; ++r)
          S[(size_t)(ti2 * 32 + sm * 16 + h * 8 + r) * SEQN + tj2 * 32 + sn * 16 + idx] = -1e30f;
    return;
  }
  const __bf16* arow0 = qb + (size_t)(ti2 * 32 + idx)      * DIMD + h * 8;
  const __bf16* arow1 = qb + (size_t)(ti2 * 32 + 16 + idx) * DIMD + h * 8;
  const __bf16* brow0 = kb + (size_t)(tj2 * 32 + idx)      * DIMD + h * 8;
  const __bf16* brow1 = kb + (size_t)(tj2 * 32 + 16 + idx) * DIMD + h * 8;
  v8f c00 = {}, c01 = {}, c10 = {}, c11 = {};
  for (int k0 = 0; k0 < DIMD; k0 += 32) {
    v16bf a0 = load_fragA(arow0 + k0);
    v16bf a1 = load_fragA(arow1 + k0);
    v16bf b0 = load_fragB(brow0 + k0);
    v16bf b1 = load_fragB(brow1 + k0);
    c00 = __builtin_amdgcn_wmma_f32_16x16x32_bf16(false, a0, false, b0, (short)0, c00, false, false);
    c01 = __builtin_amdgcn_wmma_f32_16x16x32_bf16(false, a0, false, b1, (short)0, c01, false, false);
    c10 = __builtin_amdgcn_wmma_f32_16x16x32_bf16(false, a1, false, b0, (short)0, c10, false, false);
    c11 = __builtin_amdgcn_wmma_f32_16x16x32_bf16(false, a1, false, b1, (short)0, c11, false, false);
  }
  const float scale = 0.03125f;                  // 1/sqrt(1024)
  const v8f* cc[4] = { &c00, &c01, &c10, &c11 };
#pragma unroll
  for (int sm = 0; sm < 2; ++sm)
#pragma unroll
    for (int sn = 0; sn < 2; ++sn) {
      const v8f& c = *cc[sm * 2 + sn];
#pragma unroll
      for (int r = 0; r < 8; ++r) {
        int row = ti2 * 32 + sm * 16 + h * 8 + r;
        int col = tj2 * 32 + sn * 16 + idx;
        S[(size_t)row * SEQN + col] = (col <= row) ? c[r] * scale : -1e30f;
      }
    }
}

// ---- causal softmax over one row, emit bf16 P (zeros above diagonal) ---
__global__ void k_softmax(const float* __restrict__ S, __bf16* __restrict__ P) {
  const int n = blockIdx.x;
  const int t = threadIdx.x;
  __shared__ float red[256];
  const float* row = S + (size_t)n * SEQN;
  const int len = n + 1;
  float m = -1e30f;
  for (int j = t; j < len; j += 256) m = fmaxf(m, row[j]);
  red[t] = m; __syncthreads();
  for (int s = 128; s > 0; s >>= 1) { if (t < s) red[t] = fmaxf(red[t], red[t + s]); __syncthreads(); }
  m = red[0]; __syncthreads();
  float sum = 0.f;
  for (int j = t; j < len; j += 256) sum += __expf(row[j] - m);
  red[t] = sum; __syncthreads();
  for (int s = 128; s > 0; s >>= 1) { if (t < s) red[t] += red[t + s]; __syncthreads(); }
  const float inv = 1.0f / red[0];
  for (int j = t; j < SEQN; j += 256) {
    float p = (j < len) ? __expf(row[j] - m) * inv : 0.0f;
    P[(size_t)n * SEQN + j] = (__bf16)p;
  }
}

// ---- PV GEMM (one batch): attn = P * V; 32x32 tile, K truncated --------
__global__ void k_pv(const __bf16* __restrict__ P, const __bf16* __restrict__ vt,
                     float* __restrict__ attn) {
  const int lane = threadIdx.x & 31;
  const int wid  = (blockIdx.x * blockDim.x + threadIdx.x) >> 5;
  const int tv2 = wid & 31, ti2 = wid >> 5;     // 64 row super-tiles x 32 v super-tiles
  const int idx = lane & 15, h = lane >> 4;
  const __bf16* arow0 = P  + (size_t)(ti2 * 32 + idx)      * SEQN + h * 8;
  const __bf16* arow1 = P  + (size_t)(ti2 * 32 + 16 + idx) * SEQN + h * 8;
  const __bf16* brow0 = vt + (size_t)(tv2 * 32 + idx)      * SEQN + h * 8;
  const __bf16* brow1 = vt + (size_t)(tv2 * 32 + 16 + idx) * SEQN + h * 8;
  const int kend = (ti2 + 1) * 32;               // P rows zero-padded past diagonal
  v8f c00 = {}, c01 = {}, c10 = {}, c11 = {};
  for (int k0 = 0; k0 < kend; k0 += 32) {
    v16bf a0 = load_fragA(arow0 + k0);
    v16bf a1 = load_fragA(arow1 + k0);
    v16bf b0 = load_fragB(brow0 + k0);
    v16bf b1 = load_fragB(brow1 + k0);
    c00 = __builtin_amdgcn_wmma_f32_16x16x32_bf16(false, a0, false, b0, (short)0, c00, false, false);
    c01 = __builtin_amdgcn_wmma_f32_16x16x32_bf16(false, a0, false, b1, (short)0, c01, false, false);
    c10 = __builtin_amdgcn_wmma_f32_16x16x32_bf16(false, a1, false, b0, (short)0, c10, false, false);
    c11 = __builtin_amdgcn_wmma_f32_16x16x32_bf16(false, a1, false, b1, (short)0, c11, false, false);
  }
  const v8f* cc[4] = { &c00, &c01, &c10, &c11 };
#pragma unroll
  for (int sm = 0; sm < 2; ++sm)
#pragma unroll
    for (int sn = 0; sn < 2; ++sn) {
      const v8f& c = *cc[sm * 2 + sn];
#pragma unroll
      for (int r = 0; r < 8; ++r)
        attn[(size_t)(ti2 * 32 + sm * 16 + h * 8 + r) * DIMD + tv2 * 32 + sn * 16 + idx] = c[r];
    }
}

// ---- residual + LayerNorm, emit bf16 for the FC GEMM -------------------
__global__ void k_addln(const float* __restrict__ attn, const float* __restrict__ vf,
                        const float* __restrict__ gamma, const float* __restrict__ beta,
                        __bf16* __restrict__ normed) {
  const int row = blockIdx.x;
  const int t = threadIdx.x;
  __shared__ float red[256];
  const float* pa = attn + (size_t)row * DIMD;
  const float* pv = vf   + (size_t)row * DIMD;
  float s = 0.f, s2 = 0.f;
  for (int j = t; j < DIMD; j += 256) {
    float hv = pa[j] + pv[j];
    s += hv; s2 += hv * hv;
  }
  red[t] = s; __syncthreads();
  for (int k = 128; k > 0; k >>= 1) { if (t < k) red[t] += red[t + k]; __syncthreads(); }
  float mean = red[0] * (1.0f / DIMD);
  __syncthreads();
  red[t] = s2; __syncthreads();
  for (int k = 128; k > 0; k >>= 1) { if (t < k) red[t] += red[t + k]; __syncthreads(); }
  float var = red[0] * (1.0f / DIMD) - mean * mean;
  float inv = rsqrtf(var + 1e-5f);
  for (int j = t; j < DIMD; j += 256) {
    float hv = pa[j] + pv[j];
    normed[(size_t)row * DIMD + j] = (__bf16)((hv - mean) * inv * gamma[j] + beta[j]);
  }
}

// ---- final FC GEMM with bias, f32 out, 32x32 tile per wave -------------
__global__ void k_fc(const __bf16* __restrict__ nb_, const __bf16* __restrict__ WfcT,
                     const float* __restrict__ bias, float* __restrict__ out) {
  const int lane = threadIdx.x & 31;
  const int wid  = (blockIdx.x * blockDim.x + threadIdx.x) >> 5;
  const int tn2 = wid & 31;   // 32 col super-tiles
  const int tm2 = wid >> 5;   // 256 row super-tiles
  const int idx = lane & 15, h = lane >> 4;
  const __bf16* arow0 = nb_  + (size_t)(tm2 * 32 + idx)      * DIMD + h * 8;
  const __bf16* arow1 = nb_  + (size_t)(tm2 * 32 + 16 + idx) * DIMD + h * 8;
  const __bf16* brow0 = WfcT + (size_t)(tn2 * 32 + idx)      * DIMD + h * 8;
  const __bf16* brow1 = WfcT + (size_t)(tn2 * 32 + 16 + idx) * DIMD + h * 8;
  v8f c00 = {}, c01 = {}, c10 = {}, c11 = {};
  for (int k0 = 0; k0 < DIMD; k0 += 32) {
    v16bf a0 = load_fragA(arow0 + k0);
    v16bf a1 = load_fragA(arow1 + k0);
    v16bf b0 = load_fragB(brow0 + k0);
    v16bf b1 = load_fragB(brow1 + k0);
    c00 = __builtin_amdgcn_wmma_f32_16x16x32_bf16(false, a0, false, b0, (short)0, c00, false, false);
    c01 = __builtin_amdgcn_wmma_f32_16x16x32_bf16(false, a0, false, b1, (short)0, c01, false, false);
    c10 = __builtin_amdgcn_wmma_f32_16x16x32_bf16(false, a1, false, b0, (short)0, c10, false, false);
    c11 = __builtin_amdgcn_wmma_f32_16x16x32_bf16(false, a1, false, b1, (short)0, c11, false, false);
  }
  const v8f* cc[4] = { &c00, &c01, &c10, &c11 };
#pragma unroll
  for (int sm = 0; sm < 2; ++sm)
#pragma unroll
    for (int sn = 0; sn < 2; ++sn) {
      const v8f& c = *cc[sm * 2 + sn];
      const int col = tn2 * 32 + sn * 16 + idx;
      const float bi = bias[col];
#pragma unroll
      for (int r = 0; r < 8; ++r)
        out[(size_t)(tm2 * 32 + sm * 16 + h * 8 + r) * DIMD + col] = c[r] + bi;
    }
}

// ------------------------------------------------------------------------
extern "C" void kernel_launch(void* const* d_in, const int* in_sizes, int n_in,
                              void* d_out, int out_size, void* d_ws, size_t ws_size,
                              hipStream_t stream) {
  const float* x     = (const float*)d_in[0];
  const float* Wq    = (const float*)d_in[1];
  const float* Wk    = (const float*)d_in[2];
  const float* Wv    = (const float*)d_in[3];
  const float* Wfc   = (const float*)d_in[4];
  const float* bfc   = (const float*)d_in[5];
  const float* gamma = (const float*)d_in[6];
  const float* beta  = (const float*)d_in[7];

  char* ws = (char*)d_ws;
  const size_t MB = 1ull << 20;
  __bf16* xb     = (__bf16*)(ws);             //  16 MB  x in bf16 [8192][1024]
  __bf16* WqT    = (__bf16*)(ws + 16  * MB);  //   2 MB
  __bf16* WkT    = (__bf16*)(ws + 18  * MB);  //   2 MB
  __bf16* WvT    = (__bf16*)(ws + 20  * MB);  //   2 MB
  __bf16* WfcT   = (__bf16*)(ws + 22  * MB);  //   2 MB
  __bf16* qb     = (__bf16*)(ws + 24  * MB);  //  16 MB
  __bf16* kb     = (__bf16*)(ws + 40  * MB);  //  16 MB
  __bf16* vt     = (__bf16*)(ws + 56  * MB);  //  16 MB  V^T per batch [4][1024][2048]
  float*  vf     = (float*) (ws + 72  * MB);  //  32 MB  V f32 (residual)
  float*  attn   = (float*) (ws + 104 * MB);  //  32 MB  att@V
  __bf16* normed = (__bf16*)(ws + 136 * MB);  //  16 MB
  float*  S      = (float*) (ws + 152 * MB);  //  16 MB  scores, reused per batch (L2-resident)
  __bf16* P      = (__bf16*)(ws + 168 * MB);  //   8 MB  probs,  reused per batch

  k_cvt_bf16<<<(NROWS * DIMD) / 256, 256, 0, stream>>>(x, xb, NROWS * DIMD);
  k_transpose_bf16<<<(DIMD * DIMD) / 256, 256, 0, stream>>>(Wq,  WqT);
  k_transpose_bf16<<<(DIMD * DIMD) / 256, 256, 0, stream>>>(Wk,  WkT);
  k_transpose_bf16<<<(DIMD * DIMD) / 256, 256, 0, stream>>>(Wv,  WvT);
  k_transpose_bf16<<<(DIMD * DIMD) / 256, 256, 0, stream>>>(Wfc, WfcT);

  // 256 row super-tiles * 64 col tiles / 8 waves-per-block
  k_qkv<<<2048, 256, 0, stream>>>(xb, WqT, WkT, WvT, qb, kb, vf, vt);

  for (int b = 0; b < NB; ++b) {
    const size_t ro = (size_t)b * SEQN * DIMD;
    k_scores <<<512, 256, 0, stream>>>(qb + ro, kb + ro, S);      // 64*64/8
    k_softmax<<<SEQN, 256, 0, stream>>>(S, P);
    k_pv     <<<256, 256, 0, stream>>>(P, vt + (size_t)b * DIMD * SEQN, attn + ro); // 64*32/8
  }

  k_addln<<<NROWS, 256, 0, stream>>>(attn, vf, gamma, beta, normed);
  k_fc<<<1024, 256, 0, stream>>>(normed, WfcT, bfc, (float*)d_out);
}
